// SupernodePooling_34943853920677
// MI455X (gfx1250) — compile-verified, matrix operand order checked by reading, not run on hardware
//
#include <hip/hip_runtime.h>
#include <hip/hip_bf16.h>

// ---------------------------------------------------------------------------
// SupernodePooling for MI455X (gfx1250, wave32, WMMA)
//
// Problem sizes (from reference): B=8, N=8192, S=1024, K=16, HIDDEN=192, NDIM=3
//
// Pipeline:
//   prep_weights : f32 w1/w2 -> f16, pre-swizzled into WMMA B-fragment order
//   knn          : per-supernode top-16 nearest neighbors -> gathered coords
//   fused MLP    : sincos embed + proj -> GEMM1(192x192) -> gelu ->
//                  GEMM2(192x192) -> mean over K=16  (one 16-row WMMA tile
//                  per supernode; pooling = column-sum of the D tile)
// ---------------------------------------------------------------------------

#define BB 8
#define NN 8192
#define SS 1024
#define KK 16
#define HH 192

typedef __attribute__((ext_vector_type(16))) _Float16 v16h;
typedef __attribute__((ext_vector_type(8)))  _Float16 v8h;
typedef __attribute__((ext_vector_type(8)))  float    v8f;

// ---------------------------------------------------------------------------
// Kernel 0: convert w (192x192 f32, [k][n]) into f16 B fragments.
// Fragment layout: [nt (12)][kt (6)][lane (32)][j (16)] halfs.
// For lane l, element j of the 32x16 (KxN) tile:
//   n = nt*16 + (l%16)
//   khalf = (l<16) ? 0 : 8
//   k = kt*32 + ( j<8 ? khalf + j : 16 + khalf + (j-8) )
// (mirrors the documented 16-bit A-matrix lane/K striping)
// ---------------------------------------------------------------------------
__global__ void snp_prep_weights(const float* __restrict__ w1,
                                 const float* __restrict__ w2,
                                 _Float16* __restrict__ wh) {
    int gid = blockIdx.x * 256 + threadIdx.x;     // 0 .. 73727
    if (gid >= 2 * 12 * 6 * 32 * 16) return;
    int which = gid / 36864;
    int idx   = gid - which * 36864;
    int j  = idx & 15;
    int l  = (idx >> 4) & 31;
    int kt = (idx >> 9) % 6;
    int nt = idx / 3072;
    int n = nt * 16 + (l & 15);
    int khalf = (l < 16) ? 0 : 8;
    int k = kt * 32 + ((j < 8) ? (khalf + j) : (16 + khalf + (j - 8)));
    const float* w = which ? w2 : w1;
    wh[gid] = (_Float16)w[k * HH + n];
}

// ---------------------------------------------------------------------------
// Kernel 1: kNN gather. One thread per supernode (8192 total).
// 256 threads/block -> 32 blocks; each block's 256 supernodes share a batch.
// Points staged in LDS in 2048-point chunks (24 KB).
// Top-16 kept unsorted with replace-max (mean-pool is permutation invariant).
// ---------------------------------------------------------------------------
__global__ void snp_knn(const float* __restrict__ pts,
                        const int*   __restrict__ sidx,
                        float*       __restrict__ nbr) {
    __shared__ float lp[2048 * 3];
    const int tid = threadIdx.x;
    const int sn  = blockIdx.x * 256 + tid;       // global supernode id
    const int b   = sn >> 10;                     // / S
    const float* bp = pts + (size_t)b * NN * 3;

    const int si = sidx[sn];
    const float sx = bp[si * 3 + 0];
    const float sy = bp[si * 3 + 1];
    const float sz = bp[si * 3 + 2];

    float bd[KK], bx[KK], by[KK], bz[KK];
#pragma unroll
    for (int i = 0; i < KK; ++i) { bd[i] = 3.0e38f; bx[i] = 0.f; by[i] = 0.f; bz[i] = 0.f; }
    float worst = 3.0e38f;

    for (int ch = 0; ch < 4; ++ch) {
        __syncthreads();
        for (int i = tid; i < 2048 * 3; i += 256)
            lp[i] = bp[ch * 2048 * 3 + i];
        __syncthreads();
        for (int p = 0; p < 2048; ++p) {
            float px = lp[p * 3 + 0];
            float py = lp[p * 3 + 1];
            float pz = lp[p * 3 + 2];
            float dx = px - sx, dy = py - sy, dz = pz - sz;
            float d = dx * dx + dy * dy + dz * dz;
            if (d < worst) {
                int im = 0; float m = bd[0];
#pragma unroll
                for (int i = 1; i < KK; ++i) if (bd[i] > m) { m = bd[i]; im = i; }
#pragma unroll
                for (int i = 0; i < KK; ++i)
                    if (i == im) { bd[i] = d; bx[i] = px; by[i] = py; bz[i] = pz; }
                worst = bd[0];
#pragma unroll
                for (int i = 1; i < KK; ++i) worst = fmaxf(worst, bd[i]);
            }
        }
    }

    float* o = nbr + (size_t)sn * KK * 3;
#pragma unroll
    for (int i = 0; i < KK; ++i) {
        o[i * 3 + 0] = bx[i];
        o[i * 3 + 1] = by[i];
        o[i * 3 + 2] = bz[i];
    }
}

// ---------------------------------------------------------------------------
// Kernel 2: fused embed + MLP + mean-pool. One wave = one supernode (16 rows).
// 128 threads/block = 4 waves; 2048 blocks.
// Per-wave LDS tiles (16 rows x 192 cols, stride 200 halfs for 16B-aligned
// fragment chunks + bank skew): X (embedded input) and H (post-gelu hidden).
// ---------------------------------------------------------------------------
#define ROWSTRIDE 200   // halfs; 400 B, multiple of 16 B

__device__ __forceinline__ v16h load_afrag(const _Float16* base, int kt, int lane) {
    // A fragment (16x32 f16): lane l holds row m = l%16;
    // K chunks: lanes 0-15 -> [kt*32 .. +7] and [kt*32+16 .. +23]
    //           lanes 16-31 -> [kt*32+8 .. +15] and [kt*32+24 .. +31]
    int m  = lane & 15;
    int c0 = kt * 32 + ((lane < 16) ? 0 : 8);
    v8h lo = *(const v8h*)(base + m * ROWSTRIDE + c0);
    v8h hi = *(const v8h*)(base + m * ROWSTRIDE + c0 + 16);
    return __builtin_shufflevector(lo, hi, 0, 1, 2, 3, 4, 5, 6, 7,
                                           8, 9, 10, 11, 12, 13, 14, 15);
}

__device__ __forceinline__ float gelu_tanh(float v) {
    // jax.nn.gelu default (approximate=True)
    return 0.5f * v * (1.0f + tanhf(0.7978845608028654f * (v + 0.044715f * v * v * v)));
}

__global__ void __launch_bounds__(128)
snp_mlp(const float*    __restrict__ nbr,
        const _Float16* __restrict__ w1h,
        const _Float16* __restrict__ w2h,
        const float*    __restrict__ w_in,
        const float*    __restrict__ b_in,
        const float*    __restrict__ b1,
        const float*    __restrict__ b2,
        float*          __restrict__ out) {
    __shared__ __align__(16) _Float16 ldsX[4][16 * ROWSTRIDE];
    __shared__ __align__(16) _Float16 ldsH[4][16 * ROWSTRIDE];

    const int lane = threadIdx.x & 31;
    const int wv   = threadIdx.x >> 5;
    const int sn   = blockIdx.x * 4 + wv;         // supernode id, 0..8191

    _Float16* X = ldsX[wv];
    _Float16* H = ldsH[wv];
    const float* rowc = nbr + (size_t)sn * KK * 3;

    // ---- Stage 1: input projection + continuous sincos embedding -> X (f16)
    // 16 rows x 192 cols = 3072 elements, 96 steps of 32 lanes.
    // channel c in [0,192): coord = c/64; within = c%64;
    //   within<32 -> sin(x*omega[within]) else cos(x*omega[within-32])
    //   omega[i] = 10000^(-i/32) = exp2(-i * log2(10000)/32)
    const float kFreq = -0.41524101186092570f;    // -log2(10000)/32
    for (int step = 0; step < 96; ++step) {
        int f   = step * 32 + lane;
        int r   = f / HH;
        int col = f - r * HH;
        int c   = col >> 6;                       // /64
        int within = col & 63;
        float x0 = rowc[r * 3 + 0];
        float x1 = rowc[r * 3 + 1];
        float x2 = rowc[r * 3 + 2];
        float proj = x0 * w_in[col] + x1 * w_in[HH + col] + x2 * w_in[2 * HH + col]
                   + b_in[col];
        float xc = (c == 0) ? x0 : ((c == 1) ? x1 : x2);
        int fi = within & 31;
        float ang = xc * exp2f(kFreq * (float)fi);
        float emb = (within < 32) ? sinf(ang) : cosf(ang);
        X[r * ROWSTRIDE + col] = (_Float16)(proj + emb);
    }
    __syncthreads();

    // ---- Stage 2: GEMM1 (16x192 @ 192x192) + bias + gelu -> H (f16)
    for (int nt = 0; nt < 12; ++nt) {
        v8f acc = {};
#pragma unroll
        for (int kt = 0; kt < 6; ++kt) {
            v16h a  = load_afrag(X, kt, lane);
            v16h bm = *(const v16h*)(w1h + ((nt * 6 + kt) * 32 + lane) * 16);
            acc = __builtin_amdgcn_wmma_f32_16x16x32_f16(
                false, a, false, bm, (short)0, acc, false, false);
        }
        // D layout: VGPR r, lanes 0-15: (M=r, N=lane); lanes 16-31: (M=8+r, N=lane-16)
        int n = nt * 16 + (lane & 15);
        float bias = b1[n];
        int mbase = (lane < 16) ? 0 : 8;
#pragma unroll
        for (int r = 0; r < 8; ++r) {
            float g = gelu_tanh(acc[r] + bias);
            H[(mbase + r) * ROWSTRIDE + n] = (_Float16)g;
        }
    }
    __syncthreads();

    // ---- Stage 3: GEMM2 + mean-pool over K=16 rows (+ b2)
    const float inv_k = 1.0f / (float)KK;
    for (int nt = 0; nt < 12; ++nt) {
        v8f acc = {};
#pragma unroll
        for (int kt = 0; kt < 6; ++kt) {
            v16h a  = load_afrag(H, kt, lane);
            v16h bm = *(const v16h*)(w2h + ((nt * 6 + kt) * 32 + lane) * 16);
            acc = __builtin_amdgcn_wmma_f32_16x16x32_f16(
                false, a, false, bm, (short)0, acc, false, false);
        }
        // column sum: per-lane over 8 M values, then add the other lane half
        float s = acc[0] + acc[1] + acc[2] + acc[3] + acc[4] + acc[5] + acc[6] + acc[7];
        s += __shfl_xor(s, 16, 32);
        int nl = lane & 15;
        if (lane < 16)
            out[(size_t)sn * HH + nt * 16 + nl] = s * inv_k + b2[nt * 16 + nl];
    }
}

// ---------------------------------------------------------------------------
// Launch
// ---------------------------------------------------------------------------
extern "C" void kernel_launch(void* const* d_in, const int* in_sizes, int n_in,
                              void* d_out, int out_size, void* d_ws, size_t ws_size,
                              hipStream_t stream) {
    const float* pts  = (const float*)d_in[0];   // (B,N,3)
    const int*   sidx = (const int*)  d_in[1];   // (B,S)
    const float* w_in = (const float*)d_in[2];   // (3,192)
    const float* b_in = (const float*)d_in[3];   // (192)
    const float* w1   = (const float*)d_in[4];   // (192,192)
    const float* b1   = (const float*)d_in[5];   // (192)
    const float* w2   = (const float*)d_in[6];   // (192,192)
    const float* b2   = (const float*)d_in[7];   // (192)
    float* out = (float*)d_out;                  // (B,S,192)

    // Workspace layout:
    //   [0, 73728)        : w1 f16 fragments (36864 halfs)
    //   [73728, 147456)   : w2 f16 fragments
    //   [147456, +1.5 MB) : gathered neighbor coords (B,S,K,3) f32
    _Float16* wh  = (_Float16*)d_ws;
    _Float16* w1h = wh;
    _Float16* w2h = wh + 36864;
    float*    nbr = (float*)((char*)d_ws + 147456);

    snp_prep_weights<<<288, 256, 0, stream>>>(w1, w2, wh);
    snp_knn<<<(BB * SS) / 256, 256, 0, stream>>>(pts, sidx, nbr);
    snp_mlp<<<(BB * SS) / 4, 128, 0, stream>>>(nbr, w1h, w2h, w_in, b_in, b1, b2, out);
}
